// GPTCosAttention_53712861004024
// MI455X (gfx1250) — matrix operand order repaired and checked by reading, not compiled, not run on hardware
//
#include <hip/hip_runtime.h>
#include <cstdint>
#include <cstddef>

// ---------------------------------------------------------------------------
// GPT-J cosine attention on MI455X (gfx1250, wave32, WMMA).
// Heavy math on v_wmma_f32_16x16x32_bf16 (fp32 accumulate); LDS staging via
// GLOBAL_LOAD_ASYNC_TO_LDS_B128 (ASYNCcnt) instead of global->VGPR->ds_store.
// ---------------------------------------------------------------------------

typedef __attribute__((ext_vector_type(16))) __bf16 v16bf;
typedef __attribute__((ext_vector_type(8)))  float  v8f;
typedef __attribute__((ext_vector_type(4)))  unsigned int u32x4;
typedef __attribute__((ext_vector_type(4)))  float  f32x4;

constexpr int NB  = 2;
constexpr int NS  = 1024;
constexpr int NE  = 2048;
constexpr int NH  = 16;
constexpr int NHD = 128;
constexpr float LN10000 = 9.210340371976184f;

union Frag {                 // one 16x32 bf16 WMMA operand (8 VGPRs / lane)
  v16bf v;
  u32x4 u[2];
  unsigned short s[16];
};

__device__ __forceinline__ v8f wmma_bf16(v16bf a, v16bf b, v8f c) {
  return __builtin_amdgcn_wmma_f32_16x16x32_bf16(
      /*neg_a=*/false, a, /*neg_b=*/false, b,
      /*c_mod=*/(short)0, c, /*reuse_a=*/false, /*reuse_b=*/false);
}

__device__ __forceinline__ unsigned short f32_to_bf16(float f) {
  unsigned u = __float_as_uint(f);
  unsigned r = 0x7FFFu + ((u >> 16) & 1u);   // round to nearest even
  return (unsigned short)((u + r) >> 16);
}

// Async copy of 16 bytes global -> LDS (per lane). GVS addressing:
// 64-bit SGPR base + 32-bit unsigned VGPR byte offset. LDS destination is the
// low 32 bits of the generic shared-pointer address (LDS aperture offset).
__device__ __forceinline__ void async_ld16(void* lds,
                                           const unsigned short* base,
                                           unsigned byte_off) {
  unsigned l = (unsigned)(uintptr_t)lds;
  asm volatile("global_load_async_to_lds_b128 %0, %1, %2"
               :: "v"(l), "v"(byte_off), "s"(base)
               : "memory");
}
__device__ __forceinline__ void wait_async() {
  asm volatile("s_wait_asynccnt 0" ::: "memory");
}

// --------------------------- fp32 -> bf16 convert ---------------------------
__global__ __launch_bounds__(256) void cvt_f32_bf16(const float* __restrict__ src,
                                                    unsigned short* __restrict__ dst,
                                                    int n) {
  int i = (blockIdx.x * 256 + threadIdx.x) * 4;
  if (i + 3 < n) {
    f32x4 f = *(const f32x4*)(src + i);
    unsigned lo = (unsigned)f32_to_bf16(f.x) | ((unsigned)f32_to_bf16(f.y) << 16);
    unsigned hi = (unsigned)f32_to_bf16(f.z) | ((unsigned)f32_to_bf16(f.w) << 16);
    *(uint2*)(dst + i) = make_uint2(lo, hi);
  }
}

// ------------------- causal valid-key counts per position -------------------
__global__ __launch_bounds__(1024) void count_kernel(const float* __restrict__ amask,
                                                     float* __restrict__ counts) {
  int b = blockIdx.x;
  int q = threadIdx.x;
  int c = 0;
  for (int k = 0; k <= q; ++k) c += (amask[b * NS + k] == 0.0f) ? 1 : 0;
  counts[b * NS + q] = (float)c;
}

// ---------------- NT GEMM: C(MxN,f32) = A(MxK,bf16) * B(NxK,bf16)^T ---------
// 128x128 tile, BK=32, 256 threads (8 waves), each wave 32x64 via 2x4 WMMA.
// K tiles staged to LDS with async-to-LDS (no VGPR round trip).
__global__ __launch_bounds__(256) void gemm_nt_bf16(const unsigned short* __restrict__ A,
                                                    const unsigned short* __restrict__ Bm,
                                                    float* __restrict__ C,
                                                    int K, int ldc) {
  __shared__ unsigned short As[128 * 32];
  __shared__ unsigned short Bs[128 * 32];

  const int bm = blockIdx.y * 128, bn = blockIdx.x * 128;
  const int tid  = threadIdx.x;
  const int wv   = tid >> 5, lane = tid & 31;
  const int m    = lane & 15, hv = lane >> 4;
  const int wm   = (wv & 3) * 32;       // 4 waves along M
  const int wn   = (wv >> 2) * 64;      // 2 waves along N

  const v8f vzero = {0.f, 0.f, 0.f, 0.f, 0.f, 0.f, 0.f, 0.f};
  v8f acc[2][4];
  for (int mi = 0; mi < 2; ++mi)
    for (int ni = 0; ni < 4; ++ni) acc[mi][ni] = vzero;

  for (int k0 = 0; k0 < K; k0 += 32) {
    __syncthreads();                                 // protect LDS from prev iter
    for (int i = tid; i < 512; i += 256) {           // 128 rows x 4 chunks of 8
      int r = i >> 2, c = (i & 3) * 8;
      async_ld16(&As[r * 32 + c], A,
                 (unsigned)(((size_t)(bm + r) * K + k0 + c) * 2));
      async_ld16(&Bs[r * 32 + c], Bm,
                 (unsigned)(((size_t)(bn + r) * K + k0 + c) * 2));
    }
    wait_async();
    __syncthreads();

    Frag af[2], bfr[4];
    for (int mi = 0; mi < 2; ++mi) {
      int row = wm + mi * 16 + m;
      af[mi].u[0] = *(const u32x4*)&As[row * 32 + hv * 8];        // K = hv*8 .. +7
      af[mi].u[1] = *(const u32x4*)&As[row * 32 + 16 + hv * 8];   // K = 16+hv*8 ..
    }
    for (int ni = 0; ni < 4; ++ni) {
      int rn = wn + ni * 16 + m;
      bfr[ni].u[0] = *(const u32x4*)&Bs[rn * 32 + hv * 8];        // K = hv*8 .. +7
      bfr[ni].u[1] = *(const u32x4*)&Bs[rn * 32 + 16 + hv * 8];
    }
    for (int mi = 0; mi < 2; ++mi)
      for (int ni = 0; ni < 4; ++ni)
        acc[mi][ni] = wmma_bf16(af[mi].v, bfr[ni].v, acc[mi][ni]);
  }

  for (int mi = 0; mi < 2; ++mi)
    for (int ni = 0; ni < 4; ++ni)
      for (int r = 0; r < 8; ++r) {
        int row = bm + wm + mi * 16 + r + hv * 8;   // C/D layout: M = r + 8*(lane/16)
        int col = bn + wn + ni * 16 + m;
        C[(size_t)row * ldc + col] = acc[mi][ni][r];
      }
}

// ------- epilogue: rotary + L2 norm (Q,K), denom scale (V), bf16 pack -------
// V is written TRANSPOSED (b,h,d,s) so the attention B-fragments are K-major.
__global__ __launch_bounds__(128) void qkv_epilogue(const float* __restrict__ QKV,
                                                    const float* __restrict__ ncst,
                                                    const float* __restrict__ amask,
                                                    const int*   __restrict__ pos,
                                                    const float* __restrict__ counts,
                                                    unsigned short* __restrict__ Qb,
                                                    unsigned short* __restrict__ Kb,
                                                    unsigned short* __restrict__ Vt) {
  __shared__ float rq[128];
  __shared__ float rk[128];

  int bsh = blockIdx.x;
  int h  = bsh % NH;
  int bs = bsh / NH;
  int b  = bs / NS, s = bs % NS;
  int d  = threadIdx.x;

  size_t base = (size_t)bs * (3 * NE) + (size_t)h * NHD + d;
  float q = QKV[base];
  float k = QKV[base + NE];
  float v = QKV[base + 2 * NE];

  // GPT-J rotary on first 64 dims (rotate_every_two); pair exchange via shfl.
  if (d < 64) {                                  // wave-uniform branch (wave32)
    int   i    = d >> 1;
    float freq = __expf(-((float)(2 * i) / 64.0f) * LN10000);
    float ang  = (float)pos[bs] * freq;
    float sn = __sinf(ang), cs = __cosf(ang);
    float qo = __shfl_xor(q, 1, 32);
    float ko = __shfl_xor(k, 1, 32);
    if ((d & 1) == 0) { q = q * cs - qo * sn; k = k * cs - ko * sn; }
    else              { q = q * cs + qo * sn; k = k * cs + ko * sn; }
  }

  rq[d] = q * q; rk[d] = k * k;
  __syncthreads();
  for (int off = 64; off > 0; off >>= 1) {
    if (d < off) { rq[d] += rq[d + off]; rk[d] += rk[d + off]; }
    __syncthreads();
  }
  float qn = sqrtf(rq[0]), kn = sqrtf(rk[0]);

  float m0    = (amask[bs] == 0.0f) ? 1.0f : 0.0f;
  float sg    = 1.0f / (1.0f + __expf(-ncst[h]));
  float denom = fmaxf(__powf(counts[bs], sg), 1.0f);

  q = q / fmaxf(qn, 1e-12f) * m0;
  k = k / fmaxf(kn, 1e-12f) * m0;
  v = v / denom * m0;

  size_t oqk = ((size_t)(b * NH + h) * NS + s) * NHD + d;
  Qb[oqk] = f32_to_bf16(q);
  Kb[oqk] = f32_to_bf16(k);
  size_t ov = ((size_t)(b * NH + h) * NHD + d) * NS + s;   // transposed
  Vt[ov] = f32_to_bf16(v);
}

// ---- attention: O = tril(Qn Kn^T) V  per (b,h); 64 q-rows / block ----------
__global__ __launch_bounds__(128) void attn_kernel(const unsigned short* __restrict__ Qb,
                                                   const unsigned short* __restrict__ Kb,
                                                   const unsigned short* __restrict__ Vt,
                                                   unsigned short* __restrict__ attnB) {
  __shared__ unsigned short Ks[32 * 128];     // K chunk: row = key,  col = d
  __shared__ unsigned short Vts[128 * 32];    // V chunk: row = d,    col = key
  __shared__ unsigned short Ss[4 * 16 * 32];  // per-wave S tile (C-layout -> A-layout)

  const int NTILE = NS / 64;                  // 16 query tiles per (b,h)
  int tile = blockIdx.x % NTILE;
  int bh   = blockIdx.x / NTILE;
  int b = bh / NH, h = bh % NH;

  const int tid = threadIdx.x, w = tid >> 5, lane = tid & 31;
  const int m = lane & 15, hv = lane >> 4;
  const int q0 = tile * 64 + w * 16;

  // Q A-fragments for the 4 K-dim chunks of HD=128 (loop-invariant).
  Frag aq[4];
  const unsigned short* qrow = Qb + ((size_t)bh * NS + q0 + m) * NHD;
  for (int kd = 0; kd < 4; ++kd) {
    aq[kd].u[0] = *(const u32x4*)(qrow + kd * 32 + hv * 8);
    aq[kd].u[1] = *(const u32x4*)(qrow + kd * 32 + 16 + hv * 8);
  }

  const v8f vzero = {0.f, 0.f, 0.f, 0.f, 0.f, 0.f, 0.f, 0.f};
  v8f o[8];
  for (int nt = 0; nt < 8; ++nt) o[nt] = vzero;

  const int nkb = 2 * tile + 2;               // uniform trip count (syncthreads-safe)
  for (int kb = 0; kb < nkb; ++kb) {
    const int k0 = kb * 32;
    __syncthreads();                           // protect Ks/Vts/Ss from prev iter
    for (int i = tid; i < 512; i += 128) {     // async-stage K (32x128) and Vt (128x32)
      int rk_ = i >> 4, ck = (i & 15) * 8;
      async_ld16(&Ks[rk_ * 128 + ck], Kb,
                 (unsigned)((((size_t)bh * NS + k0 + rk_) * NHD + ck) * 2));
      int rv = i >> 2, cv = (i & 3) * 8;
      async_ld16(&Vts[rv * 32 + cv], Vt,
                 (unsigned)((((size_t)bh * NHD + rv) * NS + k0 + cv) * 2));
    }
    wait_async();
    __syncthreads();

    // S = Q K^T for keys [k0,k0+32): two 16x16 fp32 tiles.
    v8f s0 = vzero, s1 = vzero;
    for (int kd = 0; kd < 4; ++kd) {
      Frag b0, b1;
      b0.u[0] = *(const u32x4*)&Ks[m * 128 + kd * 32 + hv * 8];
      b0.u[1] = *(const u32x4*)&Ks[m * 128 + kd * 32 + 16 + hv * 8];
      b1.u[0] = *(const u32x4*)&Ks[(m + 16) * 128 + kd * 32 + hv * 8];
      b1.u[1] = *(const u32x4*)&Ks[(m + 16) * 128 + kd * 32 + 16 + hv * 8];
      s0 = wmma_bf16(aq[kd].v, b0.v, s0);
      s1 = wmma_bf16(aq[kd].v, b1.v, s1);
    }

    // Causal mask (zero, not -inf: no softmax here), pack bf16 into LDS tile.
    for (int r = 0; r < 8; ++r) {
      int qq = q0 + r + hv * 8;
      float e0 = (k0 + m      <= qq) ? s0[r] : 0.0f;
      float e1 = (k0 + 16 + m <= qq) ? s1[r] : 0.0f;
      Ss[w * 512 + (r + hv * 8) * 32 + m]      = f32_to_bf16(e0);
      Ss[w * 512 + (r + hv * 8) * 32 + 16 + m] = f32_to_bf16(e1);
    }
    __syncthreads();                           // cross-lane C->A relayout

    Frag as;
    as.u[0] = *(const u32x4*)&Ss[w * 512 + m * 32 + hv * 8];
    as.u[1] = *(const u32x4*)&Ss[w * 512 + m * 32 + 16 + hv * 8];

    for (int nt = 0; nt < 8; ++nt) {           // O += S * V  (HD split in 8 N-tiles)
      Frag bv;                                 // K-major thanks to transposed V
      bv.u[0] = *(const u32x4*)&Vts[(nt * 16 + m) * 32 + hv * 8];
      bv.u[1] = *(const u32x4*)&Vts[(nt * 16 + m) * 32 + 16 + hv * 8];
      o[nt] = wmma_bf16(as.v, bv.v, o[nt]);
    }
  }

  // Store O as bf16 in (b*S, E) layout for the output projection GEMM.
  for (int nt = 0; nt < 8; ++nt)
    for (int r = 0; r < 8; ++r) {
      int qq = q0 + r + hv * 8;
      attnB[((size_t)(b * NS + qq)) * NE + h * NHD + nt * 16 + m] = f32_to_bf16(o[nt][r]);
    }
}

// ---------------------------------------------------------------------------
extern "C" void kernel_launch(void* const* d_in, const int* in_sizes, int n_in,
                              void* d_out, int out_size, void* d_ws, size_t ws_size,
                              hipStream_t stream) {
  (void)in_sizes; (void)n_in; (void)out_size; (void)ws_size;

  const float* hs    = (const float*)d_in[0];
  const float* w_q   = (const float*)d_in[1];
  const float* w_k   = (const float*)d_in[2];
  const float* w_v   = (const float*)d_in[3];
  const float* w_o   = (const float*)d_in[4];
  const float* ncst  = (const float*)d_in[5];
  const float* amask = (const float*)d_in[6];
  const int*   pos   = (const int*)d_in[7];
  float*       out   = (float*)d_out;

  uint8_t* ws = (uint8_t*)d_ws;
  constexpr size_t MB = 1u << 20;
  unsigned short* Xb    = (unsigned short*)(ws + 0);        //  8 MB hidden bf16
  unsigned short* Wqkv  = (unsigned short*)(ws + 8  * MB);  // 24 MB [Wq|Wk|Wv] bf16
  unsigned short* Wo    = (unsigned short*)(ws + 32 * MB);  //  8 MB
  float*          QKV   = (float*)(ws + 40 * MB);           // 48 MB (2048 x 6144 f32)
  unsigned short* Qb    = (unsigned short*)(ws + 88  * MB); //  8 MB (b,h,s,d) bf16
  unsigned short* Kb    = (unsigned short*)(ws + 96  * MB); //  8 MB (b,h,s,d) bf16
  unsigned short* Vt    = (unsigned short*)(ws + 104 * MB); //  8 MB (b,h,d,s) bf16
  unsigned short* attnB = (unsigned short*)(ws + 112 * MB); //  8 MB
  float*          cnts  = (float*)(ws + 120 * MB);          //  8 KB

  const int NELT = NE * NE;                                 // 4 Mi, == B*S*E too
  const int cvtg = NELT / 1024;

  cvt_f32_bf16<<<cvtg, 256, 0, stream>>>(hs,  Xb,            NELT);
  cvt_f32_bf16<<<cvtg, 256, 0, stream>>>(w_q, Wqkv,          NELT);
  cvt_f32_bf16<<<cvtg, 256, 0, stream>>>(w_k, Wqkv + NELT,   NELT);
  cvt_f32_bf16<<<cvtg, 256, 0, stream>>>(w_v, Wqkv + 2*NELT, NELT);
  cvt_f32_bf16<<<cvtg, 256, 0, stream>>>(w_o, Wo,            NELT);

  count_kernel<<<NB, NS, 0, stream>>>(amask, cnts);

  // Fused QKV projection: (B*S x E) x (3E x E)^T  ->  B*S x 3E (f32)
  dim3 gqkv(3 * NE / 128, (NB * NS) / 128);
  gemm_nt_bf16<<<gqkv, 256, 0, stream>>>(Xb, Wqkv, QKV, NE, 3 * NE);

  qkv_epilogue<<<NB * NS * NH, 128, 0, stream>>>(QKV, ncst, amask, pos, cnts, Qb, Kb, Vt);

  attn_kernel<<<NB * NH * (NS / 64), 128, 0, stream>>>(Qb, Kb, Vt, attnB);

  // Output projection: (B*S x E) x (E x E)^T -> fp32 out
  dim3 gout(NE / 128, (NB * NS) / 128);
  gemm_nt_bf16<<<gout, 256, 0, stream>>>(attnB, Wo, out, NE, NE);
}
